// FastGaussian2D_1537598292676
// MI455X (gfx1250) — matrix operand order repaired and checked by reading, not compiled, not run on hardware
//
#include <hip/hip_runtime.h>
#include <math.h>

#define IMG_W 256
#define NPIX  65536     // 256*256
#define NG    2048
#define NCHUNK (NG / 32)
#define TWO_PI_F 6.283185307179586f
#define LOG2E_F  1.4426950408889634f

typedef __attribute__((ext_vector_type(16))) _Float16 v16h;
typedef __attribute__((ext_vector_type(8)))  float    v8f;
typedef __attribute__((ext_vector_type(4)))  int      v4i;
typedef __attribute__((address_space(1))) v4i* gv4i_ptr;  // global int4*
typedef __attribute__((address_space(3))) v4i* lv4i_ptr;  // LDS int4*

#if __has_builtin(__builtin_amdgcn_global_load_async_to_lds_b128) && \
    __has_builtin(__builtin_amdgcn_s_wait_asynccnt)
#define USE_ASYNC_LDS 1
#else
#define USE_ASYNC_LDS 0
#endif

#if __has_builtin(__builtin_amdgcn_exp2f)
#define EXP2F(x) __builtin_amdgcn_exp2f(x)
#else
#define EXP2F(x) exp2f(x)
#endif

// ---------------------------------------------------------------------------
// Kernel 1: per-gaussian exponent coefficients, pre-scaled by log2(e) so the
// hot loop uses a bare v_exp_f32:
//   alpha[p,n] = exp2(A + Bx*x + Cx*x^2 + By*y + Cy*y^2)
// 8 floats per gaussian (5 used; stride 8 keeps b128 alignment) -> gtab.
// ---------------------------------------------------------------------------
__global__ __launch_bounds__(256) void gauss_pre(
    const float* __restrict__ pos,   // (N,2)
    const float* __restrict__ ls,    // (N,2) log_scales
    const float* __restrict__ lo,    // (N,1) log_opacities
    float* __restrict__ gtab)        // (N,8)
{
    int n = blockIdx.x * blockDim.x + threadIdx.x;
    if (n >= NG) return;
    float px = pos[2*n+0], py = pos[2*n+1];
    float sx = fmaxf(__expf(ls[2*n+0]), 0.1f);
    float sy = fmaxf(__expf(ls[2*n+1]), 0.1f);
    float isx2 = 1.0f / (sx*sx);
    float isy2 = 1.0f / (sy*sy);
    float logw = lo[n] - __logf(TWO_PI_F) - __logf(sx) - __logf(sy);
    float* o = gtab + n*8;
    o[0] = (logw - 0.5f*(px*px*isx2 + py*py*isy2)) * LOG2E_F;  // A
    o[1] = (px*isx2) * LOG2E_F;                                 // Bx
    o[2] = (-0.5f*isx2) * LOG2E_F;                              // Cx
    o[3] = (py*isy2) * LOG2E_F;                                 // By
    o[4] = (-0.5f*isy2) * LOG2E_F;                              // Cy
    o[5] = 0.0f; o[6] = 0.0f; o[7] = 0.0f;
}

// ---------------------------------------------------------------------------
// Kernel 1b: pack B operands (colors | 1.0 | zeros) in the exact per-lane
// WMMA 32x16 f16 B-matrix register layout, one 32-byte slice per
// (chunk, lane).  Removes all divergence / rebuild cost from the hot loop.
// ---------------------------------------------------------------------------
__global__ __launch_bounds__(64) void gauss_packB(
    const float* __restrict__ col,   // (N,3)
    float* __restrict__ btab)        // (NCHUNK*32, 8) dwords of packed f16
{
    int t = blockIdx.x * blockDim.x + threadIdx.x;   // chunk*32 + lane
    int lane = t & 31;
    int c    = t >> 5;
    int colc = lane & 15;
    int bofs = (lane >> 4) << 3;
    v16h b;
    #pragma unroll
    for (int ei = 0; ei < 16; ++ei) {
        int g = c*32 + ei + (ei & 8) + bofs;         // B row = gaussian
        float wv = (colc < 3) ? col[3*g + colc] : (colc == 3 ? 1.0f : 0.0f);
        b[ei] = (_Float16)wv;
    }
    *(v16h*)(btab + t*8) = b;
}

// ---------------------------------------------------------------------------
// Kernel 2: each wave owns TWO 16-pixel tiles (one coefficient read feeds two
// exponent evals + two accumulators -> half the DS traffic per pixel).
// VALU: quadratic + v_exp_f32 directly into WMMA A layout.
// Matrix pipe: v_wmma_f32_16x16x32_f16 accumulates [r,g,b,1] sums.
// B comes from global (coalesced, cache-resident) to keep the DS port free
// for coefficients.
// ---------------------------------------------------------------------------
__global__ __launch_bounds__(256) void gauss_main(
    const float* __restrict__ gtab,  // (N,8) coefficients
    const float* __restrict__ btab,  // packed B operands
    float* __restrict__ out)         // (P,3)
{
    __shared__ float lds[NG * 8];          // 64 KB coefficient table
    __shared__ float stage[8][2][16][4];   // 4 KB output descatter

    // ---- 64 KB coefficient fill ----
#if USE_ASYNC_LDS
    {
        const char* src = (const char*)gtab;
        char* dst = (char*)lds;
        for (int i = threadIdx.x; i < (NG * 32) / 16; i += 256) {
            __builtin_amdgcn_global_load_async_to_lds_b128(
                (gv4i_ptr)(src + i * 16),
                (lv4i_ptr)(dst + i * 16),
                0, 0);
        }
        __builtin_amdgcn_s_wait_asynccnt(0);
    }
#else
    {
        const float4* src = (const float4*)gtab;
        float4* dst = (float4*)lds;
        for (int i = threadIdx.x; i < NG * 2; i += 256) dst[i] = src[i];
    }
#endif
    __syncthreads();

    const int lane = threadIdx.x & 31;
    const int w    = threadIdx.x >> 5;
    const int pair = blockIdx.x * 8 + w;     // pair of 16-pixel tiles
    const int m    = lane & 15;
    const int p0   = pair * 32 + m;          // tile 0 pixel
    const int p1   = p0 + 16;                // tile 1 pixel (same image row)
    const float xf0 = (float)(p0 & (IMG_W - 1));
    const float xf1 = (float)(p1 & (IMG_W - 1));
    const float yf  = (float)(p0 >> 8);
    const float x20 = xf0 * xf0, x21 = xf1 * xf1, y2 = yf * yf;
    const int hi   = lane >> 4;
    const int kofs = hi << 3;                // A-matrix K offset per lane half
    const int colc = lane & 15;              // B/C column

    v8f acc0 = {}, acc1 = {};
    for (int c = 0; c < NCHUNK; ++c) {
        const int base = c * 32;
        // B operand: one 32-byte vector load, exact lane layout, no logic
        const v16h b = *(const v16h*)(btab + (base + lane) * 8);
        v16h a0, a1;
        #pragma unroll
        for (int h = 0; h < 16; ++h) {
            const int k = base + h + (h & 8) + kofs;     // ISA A-layout K map
            const float4 c4 = *(const float4*)(lds + k * 8);  // ds_load_b128
            const float  c5 = lds[k * 8 + 4];                 // ds_load_b32
            float e0 = c4.x + c4.y * xf0 + c4.z * x20 + c4.w * yf + c5 * y2;
            float e1 = c4.x + c4.y * xf1 + c4.z * x21 + c4.w * yf + c5 * y2;
            a0[h] = (_Float16)EXP2F(e0);
            a1[h] = (_Float16)EXP2F(e1);
        }
        acc0 = __builtin_amdgcn_wmma_f32_16x16x32_f16(
            false, a0, false, b, (short)0, acc0, false, false);
        acc1 = __builtin_amdgcn_wmma_f32_16x16x32_f16(
            false, a1, false, b, (short)0, acc1, false, false);
    }

    // C layout: VGPR r -> M = r + 8*hi, N = colc. Descatter via LDS.
    if (colc < 4) {
        #pragma unroll
        for (int r = 0; r < 8; ++r) {
            stage[w][0][r + (hi << 3)][colc] = acc0[r];
            stage[w][1][r + (hi << 3)][colc] = acc1[r];
        }
    }
    __syncthreads();

    // lanes 0..15 -> tile 0, lanes 16..31 -> tile 1; p = pair*32 + lane
    {
        const int t  = lane >> 4;
        const int mm = lane & 15;
        const float rs = stage[w][t][mm][0];
        const float gs = stage[w][t][mm][1];
        const float bs = stage[w][t][mm][2];
        const float inv = 1.0f / fmaxf(stage[w][t][mm][3], 1e-8f);
        const int p = pair * 32 + lane;
        out[3*p+0] = fminf(fmaxf(rs * inv, 0.0f), 1.0f);
        out[3*p+1] = fminf(fmaxf(gs * inv, 0.0f), 1.0f);
        out[3*p+2] = fminf(fmaxf(bs * inv, 0.0f), 1.0f);
    }
}

// ---------------------------------------------------------------------------
extern "C" void kernel_launch(void* const* d_in, const int* in_sizes, int n_in,
                              void* d_out, int out_size, void* d_ws, size_t ws_size,
                              hipStream_t stream) {
    (void)in_sizes; (void)n_in; (void)out_size; (void)ws_size;
    // setup_inputs order: coords(0), positions(1), log_scales(2), colors(3), log_opacities(4)
    const float* pos = (const float*)d_in[1];
    const float* ls  = (const float*)d_in[2];
    const float* col = (const float*)d_in[3];
    const float* lo  = (const float*)d_in[4];
    float* gtab = (float*)d_ws;                       // 64 KB
    float* btab = (float*)d_ws + NG * 8;              // 64 KB
    float* out  = (float*)d_out;

    gauss_pre<<<NG / 256, 256, 0, stream>>>(pos, ls, lo, gtab);
    gauss_packB<<<NCHUNK * 32 / 64, 64, 0, stream>>>(col, btab);
    // 2048 tile-pairs (32 pixels each), 8 waves per 256-thread block
    gauss_main<<<(NPIX / 32) / 8, 256, 0, stream>>>(gtab, btab, out);
}
